// observation_processing_network_85255100825937
// MI455X (gfx1250) — matrix operand order repaired and checked by reading, not compiled, not run on hardware
//
#include <hip/hip_runtime.h>
#include <math.h>

// ---------------------------------------------------------------------------
// Fused observation-processing GNN for MI455X (gfx1250, wave32).
// One persistent workgroup, all state in LDS. The 4 non-trivial matmuls run
// on v_wmma_f32_16x16x32_f16 with branchless, zero-padded f16 operand tiles
// staged in LDS (contiguous per-lane fragments -> wide ds loads, no exec
// juggling). 50x50 eigh via parallel cyclic Jacobi.
// ---------------------------------------------------------------------------

typedef __attribute__((ext_vector_type(16))) _Float16 v16h;
typedef __attribute__((ext_vector_type(8)))  float    v8f;

constexpr int   NN        = 50;    // nodes
constexpr int   EE        = 400;   // directed edges (no self loops)
constexpr int   ET        = 450;   // edges + self loops
constexpr int   LD        = 51;    // padded row stride for Jacobi tiles
constexpr int   BSZ       = 256;   // 8 waves of 32
constexpr int   NW        = BSZ / 32;
constexpr float NEG_SLOPE = 0.2f;
constexpr float INV_SQRT5 = 0.44721359549995794f;

// C(64x64, ld=64) = A(64xKA) * Bt(64xKA)^T using f16 WMMA, KA = 32*Ksteps.
// Operands are pre-staged zero-padded in LDS; all accesses unconditional.
__device__ __forceinline__ void wmma_gemm(int lane, int wave,
                                          int Mtiles, int Ntiles, int Ksteps,
                                          const _Float16* __restrict__ A,
                                          const _Float16* __restrict__ Bt,
                                          float* __restrict__ C)
{
  const int m15 = lane & 15;
  const int hi  = lane >> 4;
  const int KA  = Ksteps * 32;
  const int tcount = Mtiles * Ntiles;
  for (int t = wave; t < tcount; t += NW) {   // uniform per wave: EXEC all 1s
    const int mt = t / Ntiles;
    const int nt = t % Ntiles;
    const _Float16* ap = A  + (mt * 16 + m15) * KA + hi * 8;
    const _Float16* bp = Bt + (nt * 16 + m15) * KA + hi * 8;
    v8f c = {};
    for (int ks = 0; ks < Ksteps; ++ks) {
      const int kb = ks * 32;
      v16h a, b;
#pragma unroll
      for (int j = 0; j < 8; ++j) {
        a[j]     = ap[kb + j];        // K = kb + hi*8 + j
        a[8 + j] = ap[kb + 16 + j];   // K = kb + hi*8 + 16 + j
      }
#pragma unroll
      for (int j = 0; j < 8; ++j) {
        b[j]     = bp[kb + j];        // K = kb + hi*8 + j, N = lane&15
        b[8 + j] = bp[kb + 16 + j];   // K = kb + hi*8 + 16 + j
      }
      c = __builtin_amdgcn_wmma_f32_16x16x32_f16(false, a, false, b,
                                                 (short)0, c, false, false);
    }
#pragma unroll
    for (int i = 0; i < 8; ++i)       // C/D: VGPR i -> M = i + hi*8
      C[(mt * 16 + hi * 8 + i) * 64 + nt * 16 + m15] = c[i];
  }
}

// PyG edge softmax (grouped by dst) via per-node scans; deterministic.
__device__ __forceinline__ void edge_softmax(int tid, const int* EDd,
                                             const float* ELOG, float* EALP,
                                             float* NMx, float* NSx)
{
  for (int n = tid; n < NN; n += BSZ) {
    float m = -1e30f;
    for (int e = 0; e < ET; ++e) if (EDd[e] == n) m = fmaxf(m, ELOG[e]);
    float s = 0.f;
    for (int e = 0; e < ET; ++e) if (EDd[e] == n) s += __expf(ELOG[e] - m);
    NMx[n] = m;
    NSx[n] = s;
  }
  __syncthreads();
  for (int e = tid; e < ET; e += BSZ)
    EALP[e] = __expf(ELOG[e] - NMx[EDd[e]]) / (NSx[EDd[e]] + 1e-16f);
  __syncthreads();
}

__global__ __launch_bounds__(BSZ, 1)
void obs_net_fused(const float* __restrict__ x,          // 50x3
                   const int*   __restrict__ edge_index, // 2x400
                   const float* __restrict__ mask,       // 50
                   const float* __restrict__ gat_w1,  const float* __restrict__ gat_as1,
                   const float* __restrict__ gat_ad1, const float* __restrict__ gat_b1,
                   const float* __restrict__ gat_w2,  const float* __restrict__ gat_as2,
                   const float* __restrict__ gat_ad2, const float* __restrict__ gat_b2,
                   const float* __restrict__ mha_in_w,  const float* __restrict__ mha_in_b,
                   const float* __restrict__ mha_out_w, const float* __restrict__ mha_out_b,
                   const float* __restrict__ tc_wq, const float* __restrict__ tc_bq,
                   const float* __restrict__ tc_wk, const float* __restrict__ tc_bk,
                   const float* __restrict__ tc_wv, const float* __restrict__ tc_bv,
                   const float* __restrict__ tc_ws, const float* __restrict__ tc_bs,
                   const float* __restrict__ mlp_w1, const float* __restrict__ mlp_b1,
                   const float* __restrict__ mlp_w2, const float* __restrict__ mlp_b2,
                   const float* __restrict__ critic_w, const float* __restrict__ critic_b,
                   float* __restrict__ out)
{
  // Arena overlays: Jacobi {Lm,Vm} (20400B) vs WMMA stage {PA,PBt} (16384B).
  __shared__ __align__(16) unsigned char arena[20416];
  float*    Lm  = (float*)arena;               // [NN*LD]
  float*    Vm  = (float*)(arena + 10208);     // [NN*LD]
  _Float16* PA  = (_Float16*)arena;            // [64*64] f16 A tiles
  _Float16* PBt = (_Float16*)(arena + 8192);   // [64*64] f16 B^T tiles

  __shared__ float PC[64 * 64];      // padded GEMM output (scores/attn/hidden/logits)
  __shared__ float XC[NN * 5];       // x_combined
  __shared__ float H1g[NN * 8];      // GAT1 features
  __shared__ float GS[NN * 8];       // GAT aggregation scratch
  __shared__ float Hc[NN * 5];       // current 5-dim features
  __shared__ float QKV[NN * 15];     // packed q|k|v
  __shared__ float TMP[NN * 5];
  __shared__ float FH[NN * 5];       // features entering MLP
  __shared__ int   ES[ET], EDd[ET];
  __shared__ float ELOG[ET], EALP[ET];
  __shared__ float SSa[NN], SDa[NN], NMx[NN], NSx[NN], DEG[NN];
  __shared__ float csC[25], csS[25];
  __shared__ int   ppA[25], qqA[25];
  __shared__ int   sel[2];
  __shared__ float red[64];

  const int tid  = threadIdx.x;
  const int lane = tid & 31;
  const int wave = tid >> 5;

  // ---- edges + self loops into LDS --------------------------------------
  for (int e = tid; e < ET; e += BSZ) {
    if (e < EE) { ES[e] = edge_index[e]; EDd[e] = edge_index[EE + e]; }
    else        { ES[e] = e - EE;        EDd[e] = e - EE; }
  }
  // ---- adjacency (in Lm) ------------------------------------------------
  for (int i = tid; i < NN * LD; i += BSZ) Lm[i] = 0.f;
  __syncthreads();
  for (int e = tid; e < EE; e += BSZ)
    atomicAdd(&Lm[ES[e] * LD + EDd[e]], 1.0f);
  __syncthreads();
  // ---- D^{-1/2} ---------------------------------------------------------
  for (int n = tid; n < NN; n += BSZ) {
    float d = 0.f;
    for (int j = 0; j < NN; ++j) d += Lm[n * LD + j];
    DEG[n] = (d > 0.f) ? rsqrtf(fmaxf(d, 1e-12f)) : 0.f;
  }
  __syncthreads();
  // ---- L = I - D^-1/2 A D^-1/2 ; V = I ---------------------------------
  for (int idx = tid; idx < NN * NN; idx += BSZ) {
    const int i = idx / NN, j = idx % NN;
    const float a = Lm[i * LD + j];
    Lm[i * LD + j] = ((i == j) ? 1.f : 0.f) - DEG[i] * a * DEG[j];
    Vm[i * LD + j] = (i == j) ? 1.f : 0.f;
  }
  __syncthreads();

  // ---- parallel cyclic Jacobi eigensolver (round-robin pair ordering) ---
  for (int sweep = 0; sweep < 10; ++sweep) {
    for (int r = 0; r < NN - 1; ++r) {
      if (tid < 25) {
        int p, q;
        if (tid == 0) { p = NN - 1; q = r; }
        else { p = (r + tid) % (NN - 1); q = (r - tid + (NN - 1)) % (NN - 1); }
        if (p > q) { const int t = p; p = q; q = t; }
        ppA[tid] = p; qqA[tid] = q;
        const float apq = Lm[p * LD + q];
        float c = 1.f, s = 0.f;
        if (fabsf(apq) > 1e-12f) {
          const float app = Lm[p * LD + p];
          const float aqq = Lm[q * LD + q];
          const float tau = (aqq - app) / (2.f * apq);
          const float t   = ((tau >= 0.f) ? 1.f : -1.f) /
                            (fabsf(tau) + sqrtf(1.f + tau * tau));
          c = 1.f / sqrtf(1.f + t * t);
          s = t * c;
        }
        csC[tid] = c; csS[tid] = s;
      }
      __syncthreads();
      for (int idx = tid; idx < 25 * NN; idx += BSZ) {   // row rotations
        const int pr = idx / NN, k = idx % NN;
        const int p = ppA[pr], q = qqA[pr];
        const float c = csC[pr], s = csS[pr];
        const float lp = Lm[p * LD + k], lq = Lm[q * LD + k];
        Lm[p * LD + k] = c * lp - s * lq;
        Lm[q * LD + k] = s * lp + c * lq;
      }
      __syncthreads();
      for (int idx = tid; idx < 25 * NN; idx += BSZ) {   // column rotations
        const int pr = idx / NN, k = idx % NN;
        const int p = ppA[pr], q = qqA[pr];
        const float c = csC[pr], s = csS[pr];
        const float lp = Lm[k * LD + p], lq = Lm[k * LD + q];
        Lm[k * LD + p] = c * lp - s * lq;
        Lm[k * LD + q] = s * lp + c * lq;
        const float vp = Vm[k * LD + p], vq = Vm[k * LD + q];
        Vm[k * LD + p] = c * vp - s * vq;
        Vm[k * LD + q] = s * vp + c * vq;
      }
      __syncthreads();
    }
  }
  // ---- two smallest eigenvalues -> eigenvector columns ------------------
  if (tid == 0) {
    int i0 = 0; float d0 = Lm[0];
    for (int i = 1; i < NN; ++i) {
      const float d = Lm[i * LD + i];
      if (d < d0) { d0 = d; i0 = i; }
    }
    int i1 = (i0 == 0) ? 1 : 0; float d1 = Lm[i1 * LD + i1];
    for (int i = 0; i < NN; ++i) {
      if (i == i0) continue;
      const float d = Lm[i * LD + i];
      if (d < d1) { d1 = d; i1 = i; }
    }
    sel[0] = i0; sel[1] = i1;
  }
  __syncthreads();
  // ---- x_combined = [x | lap_ev] ---------------------------------------
  for (int n = tid; n < NN; n += BSZ) {
    XC[n * 5 + 0] = x[n * 3 + 0];
    XC[n * 5 + 1] = x[n * 3 + 1];
    XC[n * 5 + 2] = x[n * 3 + 2];
    XC[n * 5 + 3] = Vm[n * LD + sel[0]];
    XC[n * 5 + 4] = Vm[n * LD + sel[1]];
  }
  __syncthreads();   // Jacobi arena dead from here; PA/PBt may reuse it

  // ======================= GAT layer 1 (5 -> 8) ==========================
  for (int idx = tid; idx < NN * 8; idx += BSZ) {
    const int n = idx >> 3, o = idx & 7;
    float acc = 0.f;
    for (int i = 0; i < 5; ++i) acc += XC[n * 5 + i] * gat_w1[o * 5 + i];
    H1g[idx] = acc;
  }
  __syncthreads();
  for (int n = tid; n < NN; n += BSZ) {
    float ss = 0.f, sd = 0.f;
    for (int o = 0; o < 8; ++o) {
      ss += H1g[n * 8 + o] * gat_as1[o];
      sd += H1g[n * 8 + o] * gat_ad1[o];
    }
    SSa[n] = ss; SDa[n] = sd;
  }
  __syncthreads();
  for (int e = tid; e < ET; e += BSZ) {
    const float z = SSa[ES[e]] + SDa[EDd[e]];
    ELOG[e] = (z > 0.f) ? z : NEG_SLOPE * z;
  }
  __syncthreads();
  edge_softmax(tid, EDd, ELOG, EALP, NMx, NSx);
  for (int idx = tid; idx < NN * 8; idx += BSZ) {
    const int n = idx >> 3, o = idx & 7;
    float acc = gat_b1[o];
    for (int e = 0; e < ET; ++e)
      if (EDd[e] == n) acc += EALP[e] * H1g[ES[e] * 8 + o];
    GS[idx] = acc;
  }
  __syncthreads();
  for (int idx = tid; idx < NN * 8; idx += BSZ) H1g[idx] = fmaxf(GS[idx], 0.f);
  __syncthreads();

  // ======================= GAT layer 2 (8 -> 5) ==========================
  for (int idx = tid; idx < NN * 5; idx += BSZ) {
    const int n = idx / 5, o = idx % 5;
    float acc = 0.f;
    for (int i = 0; i < 8; ++i) acc += H1g[n * 8 + i] * gat_w2[o * 8 + i];
    TMP[idx] = acc;
  }
  __syncthreads();
  for (int n = tid; n < NN; n += BSZ) {
    float ss = 0.f, sd = 0.f;
    for (int o = 0; o < 5; ++o) {
      ss += TMP[n * 5 + o] * gat_as2[o];
      sd += TMP[n * 5 + o] * gat_ad2[o];
    }
    SSa[n] = ss; SDa[n] = sd;
  }
  __syncthreads();
  for (int e = tid; e < ET; e += BSZ) {
    const float z = SSa[ES[e]] + SDa[EDd[e]];
    ELOG[e] = (z > 0.f) ? z : NEG_SLOPE * z;
  }
  __syncthreads();
  edge_softmax(tid, EDd, ELOG, EALP, NMx, NSx);
  for (int idx = tid; idx < NN * 5; idx += BSZ) {
    const int n = idx / 5, o = idx % 5;
    float acc = gat_b2[o];
    for (int e = 0; e < ET; ++e)
      if (EDd[e] == n) acc += EALP[e] * TMP[ES[e] * 5 + o];
    Hc[idx] = acc;
  }
  __syncthreads();

  // ======================= MultiheadAttention ============================
  for (int idx = tid; idx < NN * 15; idx += BSZ) {
    const int n = idx / 15, o = idx % 15;
    float acc = mha_in_b[o];
    for (int i = 0; i < 5; ++i) acc += Hc[n * 5 + i] * mha_in_w[o * 5 + i];
    QKV[idx] = acc;
  }
  __syncthreads();
  // scores = (q/sqrt(5)) @ k^T : stage A=q (50x5->64x32), Bt=k (n-major)
  for (int i = tid; i < 64 * 32; i += BSZ) { PA[i] = (_Float16)0.f; PBt[i] = (_Float16)0.f; }
  __syncthreads();
  for (int idx = tid; idx < NN * 5; idx += BSZ) {
    const int n = idx / 5, k = idx % 5;
    PA[n * 32 + k]  = (_Float16)(QKV[n * 15 + k] * INV_SQRT5);
    PBt[n * 32 + k] = (_Float16)(QKV[n * 15 + 5 + k]);
  }
  __syncthreads();
  wmma_gemm(lane, wave, 4, 4, 1, PA, PBt, PC);    // PC pad region exact zeros
  __syncthreads();
  for (int n = tid; n < NN; n += BSZ) {           // row softmax on valid 50x50
    float m = -1e30f;
    for (int j = 0; j < NN; ++j) m = fmaxf(m, PC[n * 64 + j]);
    float s = 0.f;
    for (int j = 0; j < NN; ++j) {
      const float e = __expf(PC[n * 64 + j] - m);
      PC[n * 64 + j] = e; s += e;
    }
    const float inv = 1.f / s;
    for (int j = 0; j < NN; ++j) PC[n * 64 + j] *= inv;
  }
  __syncthreads();
  // attn @ v : stage A=attn (64x64, pad already 0), Bt=v^T (5x50 -> 64x64)
  for (int i = tid; i < 64 * 64; i += BSZ) {
    PA[i]  = (_Float16)PC[i];
    PBt[i] = (_Float16)0.f;
  }
  __syncthreads();
  for (int idx = tid; idx < NN * 5; idx += BSZ) {
    const int k = idx / 5, n = idx % 5;
    PBt[n * 64 + k] = (_Float16)QKV[k * 15 + 10 + n];
  }
  __syncthreads();
  wmma_gemm(lane, wave, 4, 1, 2, PA, PBt, PC);
  __syncthreads();
  for (int idx = tid; idx < NN * 5; idx += BSZ) {
    const int n = idx / 5, o = idx % 5;
    float acc = mha_out_b[o];
    for (int i = 0; i < 5; ++i) acc += PC[n * 64 + i] * mha_out_w[o * 5 + i];
    Hc[idx] = acc;
  }
  __syncthreads();

  // ======================= TransformerConv ===============================
  for (int idx = tid; idx < NN * 15; idx += BSZ) {
    const int n = idx / 15, c = idx % 15;
    const float* w; const float* b; int o;
    if (c < 5)       { w = tc_wq; b = tc_bq; o = c; }
    else if (c < 10) { w = tc_wk; b = tc_bk; o = c - 5; }
    else             { w = tc_wv; b = tc_bv; o = c - 10; }
    float acc = b[o];
    for (int i = 0; i < 5; ++i) acc += Hc[n * 5 + i] * w[o * 5 + i];
    QKV[idx] = acc;
  }
  __syncthreads();
  for (int e = tid; e < ET; e += BSZ) {
    float acc = 0.f;
    for (int o = 0; o < 5; ++o)
      acc += QKV[EDd[e] * 15 + o] * QKV[ES[e] * 15 + 5 + o];
    ELOG[e] = acc * INV_SQRT5;
  }
  __syncthreads();
  edge_softmax(tid, EDd, ELOG, EALP, NMx, NSx);
  for (int idx = tid; idx < NN * 5; idx += BSZ) {
    const int n = idx / 5, o = idx % 5;
    float acc = 0.f;
    for (int e = 0; e < ET; ++e)
      if (EDd[e] == n) acc += EALP[e] * QKV[ES[e] * 15 + 10 + o];
    TMP[idx] = acc;
  }
  __syncthreads();
  for (int idx = tid; idx < NN * 5; idx += BSZ) {
    const int n = idx / 5, o = idx % 5;
    float acc = TMP[idx] + tc_bs[o];
    for (int i = 0; i < 5; ++i) acc += Hc[n * 5 + i] * tc_ws[o * 5 + i];
    FH[idx] = acc;
  }
  __syncthreads();

  // ======================= MLP head ======================================
  // padded-flat @ mlp_w1^T collapses to FH(50x5) @ w1[:, :5]^T  (K=5)
  for (int i = tid; i < 64 * 32; i += BSZ) { PA[i] = (_Float16)0.f; PBt[i] = (_Float16)0.f; }
  __syncthreads();
  for (int idx = tid; idx < NN * 5; idx += BSZ)
    PA[(idx / 5) * 32 + (idx % 5)] = (_Float16)FH[idx];
  for (int idx = tid; idx < 64 * 5; idx += BSZ)
    PBt[(idx / 5) * 32 + (idx % 5)] = (_Float16)mlp_w1[(idx / 5) * 250 + (idx % 5)];
  __syncthreads();
  wmma_gemm(lane, wave, 4, 4, 1, PA, PBt, PC);
  __syncthreads();
  for (int idx = tid; idx < NN * 64; idx += BSZ)
    PC[idx] = fmaxf(PC[idx] + mlp_b1[idx & 63], 0.f);
  __syncthreads();
  // logits = hidden(50x64) @ mlp_w2^T  (K=64 -> 2 WMMA K-steps)
  for (int i = tid; i < 64 * 64; i += BSZ) {
    PA[i]  = (_Float16)PC[i];          // rows >=50 are exact zeros already
    PBt[i] = (_Float16)0.f;
  }
  __syncthreads();
  for (int idx = tid; idx < NN * 64; idx += BSZ)
    PBt[idx] = (_Float16)mlp_w2[idx];  // (50,64) row-major == n-major, stride 64
  __syncthreads();
  wmma_gemm(lane, wave, 4, 4, 2, PA, PBt, PC);
  __syncthreads();
  for (int idx = tid; idx < NN * NN; idx += BSZ)
    PC[(idx / NN) * 64 + (idx % NN)] += mlp_b2[idx % NN];
  __syncthreads();

  // ======================= outputs =======================================
  for (int idx = tid; idx < NN * NN; idx += BSZ)
    out[idx] = PC[(idx / NN) * 64 + (idx % NN)] * mask[idx % NN];
  if (tid < NN) {
    float rv = critic_b[0];
    for (int j = 0; j < NN; ++j) rv += PC[tid * 64 + j] * critic_w[j];
    red[tid] = rv;
  }
  __syncthreads();
  if (tid == 0) {
    float s = 0.f;
    for (int i = 0; i < NN; ++i) s += red[i];
    out[NN * NN] = s / (float)NN;
  }
  for (int idx = tid; idx < NN * 5; idx += BSZ)
    out[NN * NN + 1 + idx] = XC[idx];
  for (int e = tid; e < ET; e += BSZ) {
    out[NN * NN + 1 + NN * 5 + e]      = (float)ES[e];
    out[NN * NN + 1 + NN * 5 + ET + e] = (float)EDd[e];
  }
}

extern "C" void kernel_launch(void* const* d_in, const int* in_sizes, int n_in,
                              void* d_out, int out_size, void* d_ws, size_t ws_size,
                              hipStream_t stream)
{
  (void)in_sizes; (void)n_in; (void)out_size; (void)d_ws; (void)ws_size;
  obs_net_fused<<<1, BSZ, 0, stream>>>(
      (const float*)d_in[0],  (const int*)d_in[1],    (const float*)d_in[2],
      (const float*)d_in[3],  (const float*)d_in[4],  (const float*)d_in[5],
      (const float*)d_in[6],  (const float*)d_in[7],  (const float*)d_in[8],
      (const float*)d_in[9],  (const float*)d_in[10], (const float*)d_in[11],
      (const float*)d_in[12], (const float*)d_in[13], (const float*)d_in[14],
      (const float*)d_in[15], (const float*)d_in[16], (const float*)d_in[17],
      (const float*)d_in[18], (const float*)d_in[19], (const float*)d_in[20],
      (const float*)d_in[21], (const float*)d_in[22], (const float*)d_in[23],
      (const float*)d_in[24], (const float*)d_in[25], (const float*)d_in[26],
      (const float*)d_in[27], (const float*)d_in[28],
      (float*)d_out);
}
